// RelativePositionCrossAttention_73375221284917
// MI455X (gfx1250) — compile-verified, matrix operand order checked by reading, not run on hardware
//
#include <hip/hip_runtime.h>
#include <math.h>

// ---------------------------------------------------------------------------
// Types for CDNA5 WMMA (wave32, 16x16x32 bf16 -> f32)
// ---------------------------------------------------------------------------
typedef __bf16 bf16_t;
typedef bf16_t v16bf __attribute__((ext_vector_type(16)));
typedef bf16_t v8bf  __attribute__((ext_vector_type(8)));
typedef float  v8f   __attribute__((ext_vector_type(8)));

#define DIM      768
#define HEADS    12
#define HD       64
#define QL       256
#define KL       1024
#define BATCH    32
#define BH       (BATCH * HEADS)          // 384
#define MQ       (BATCH * QL)             // 8192
#define MKV      (BATCH * KL)             // 32768
#define NKV      (2 * DIM)                // 1536
#define BIAS_QK  (QL * KL)                // 262144

// K-index of halfword j inside a 16-bit A/B fragment (ISA 7.12.2):
// lanes 0-15 hold K {0-7,16-23}, lanes 16-31 hold K {8-15,24-31}.
__device__ __forceinline__ int kof(int lane, int j) {
    return j + 8 * ((j >> 3) + (lane >> 4));
}

// Load a 16x32 bf16 A-fragment (or column-major B-fragment) where the K axis
// is contiguous in memory for each row: two 16-byte loads per lane.
__device__ __forceinline__ v16bf load_frag(const bf16_t* __restrict__ base,
                                           int row, int ldk, int kk, int lane) {
    const bf16_t* p = base + (size_t)row * ldk + kk + 8 * (lane >> 4);
    v8bf lo = *(const v8bf*)(p);
    v8bf hi = *(const v8bf*)(p + 16);
    v16bf r;
#pragma unroll
    for (int i = 0; i < 8; ++i) { r[i] = lo[i]; r[8 + i] = hi[i]; }
    return r;
}

__device__ __forceinline__ v8f wmma_bf16(v16bf a, v16bf b, v8f c) {
    return __builtin_amdgcn_wmma_f32_16x16x32_bf16(
        /*neg_a=*/false, a, /*neg_b=*/false, b,
        /*c_mod=*/(short)0, c, /*reuse_a=*/false, /*reuse_b=*/false);
}

// Accumulate a 32x64 output tile: A row-major (M x K), Wt row-major (N x K).
// Two A-fragments share every B-fragment -> 8 WMMAs per 4 B loads.
__device__ __forceinline__ void gemm32x64(const bf16_t* __restrict__ A,
                                          const bf16_t* __restrict__ Wt,
                                          int K, int m0, int n0, int lane,
                                          v8f acc[2][4]) {
    const int l16 = lane & 15;
    const bf16_t* arow0 = A + (size_t)(m0 + l16) * K + 8 * (lane >> 4);
    const bf16_t* arow1 = arow0 + (size_t)16 * K;
    for (int kk = 0; kk < K; kk += 32) {
        if (kk + 32 < K) {  // gfx1250 global_prefetch_b8 on next K-block
            __builtin_prefetch(arow0 + kk + 32, 0, 1);
            __builtin_prefetch(arow1 + kk + 32, 0, 1);
        }
        v16bf a0 = load_frag(A, m0 + l16, K, kk, lane);
        v16bf a1 = load_frag(A, m0 + 16 + l16, K, kk, lane);
#pragma unroll
        for (int t = 0; t < 4; ++t) {
            v16bf b = load_frag(Wt, n0 + t * 16 + l16, K, kk, lane);
            acc[0][t] = wmma_bf16(a0, b, acc[0][t]);
            acc[1][t] = wmma_bf16(a1, b, acc[1][t]);
        }
    }
}

__device__ __forceinline__ void zero_acc(v8f acc[2][4]) {
#pragma unroll
    for (int mi = 0; mi < 2; ++mi)
#pragma unroll
        for (int t = 0; t < 4; ++t)
#pragma unroll
            for (int r = 0; r < 8; ++r) acc[mi][t][r] = 0.0f;
}

__device__ __forceinline__ float redmax16(float v) {
    v = fmaxf(v, __shfl_xor(v, 1, 32));
    v = fmaxf(v, __shfl_xor(v, 2, 32));
    v = fmaxf(v, __shfl_xor(v, 4, 32));
    v = fmaxf(v, __shfl_xor(v, 8, 32));
    return v;
}
__device__ __forceinline__ float redsum16(float v) {
    v += __shfl_xor(v, 1, 32);
    v += __shfl_xor(v, 2, 32);
    v += __shfl_xor(v, 4, 32);
    v += __shfl_xor(v, 8, 32);
    return v;
}

// ---------------------------------------------------------------------------
// Prep kernels
// ---------------------------------------------------------------------------
__global__ void cvt_bf16_kernel(const float* __restrict__ src,
                                bf16_t* __restrict__ dst, int n) {
    int i = blockIdx.x * 256 + threadIdx.x;
    if (i < n) dst[i] = (bf16_t)src[i];
}

// src: K x N row-major (f32) -> dst: N x K row-major (bf16)
__global__ void transpose_bf16_kernel(const float* __restrict__ src,
                                      bf16_t* __restrict__ dst, int K, int N) {
    int i = blockIdx.x * 256 + threadIdx.x;
    if (i < K * N) {
        int k = i / N, n = i - k * N;
        dst[(size_t)n * K + k] = (bf16_t)src[i];
    }
}

// biasF[h][q][k] = table[rel[q*KL+k]*HEADS + h]
__global__ void bias_expand_kernel(const int* __restrict__ rel,
                                   const float* __restrict__ table,
                                   float* __restrict__ biasF) {
    int i = blockIdx.x * 256 + threadIdx.x;
    if (i < BIAS_QK) {
        int r = rel[i];
#pragma unroll
        for (int h = 0; h < HEADS; ++h)
            biasF[(size_t)h * BIAS_QK + i] = table[r * HEADS + h];
    }
}

// ---------------------------------------------------------------------------
// GEMM 1: Q = (q @ Wq + bq) * SCALE   -> Qs[b,h,256,64] bf16
// ---------------------------------------------------------------------------
__global__ __launch_bounds__(256) void qproj_kernel(
    const bf16_t* __restrict__ A, const bf16_t* __restrict__ Wt,
    const float* __restrict__ bias, bf16_t* __restrict__ Qs) {
    const int lane = threadIdx.x & 31, wid = threadIdx.x >> 5;
    const int tile = blockIdx.x * 8 + wid;
    const int NT = DIM / 64;                // 12
    const int mt = tile / NT, nt = tile - mt * NT;
    if (mt >= MQ / 32) return;
    v8f acc[2][4];
    zero_acc(acc);
    gemm32x64(A, Wt, DIM, mt * 32, nt * 64, lane, acc);
    const int half = lane >> 4, l16 = lane & 15;
#pragma unroll
    for (int mi = 0; mi < 2; ++mi)
#pragma unroll
        for (int t = 0; t < 4; ++t) {
            const int n = nt * 64 + t * 16 + l16;
            const int h = n >> 6, d = n & 63;
#pragma unroll
            for (int r = 0; r < 8; ++r) {
                const int mrow = mt * 32 + mi * 16 + 8 * half + r;
                const int b = mrow >> 8, qr = mrow & 255;
                float v = (acc[mi][t][r] + bias[n]) * 0.125f;  // SCALE=64^-0.5
                Qs[(((size_t)b * HEADS + h) * QL + qr) * HD + d] = (bf16_t)v;
            }
        }
}

// ---------------------------------------------------------------------------
// GEMM 2: KV = kv @ Wkv + bkv  -> Kb[b,h,1024,64], Vt[b,h,64,1024] (bf16)
// ---------------------------------------------------------------------------
__global__ __launch_bounds__(256) void kvproj_kernel(
    const bf16_t* __restrict__ A, const bf16_t* __restrict__ Wt,
    const float* __restrict__ bias, bf16_t* __restrict__ Kb,
    bf16_t* __restrict__ Vt) {
    const int lane = threadIdx.x & 31, wid = threadIdx.x >> 5;
    const int tile = blockIdx.x * 8 + wid;
    const int NT = NKV / 64;                // 24
    const int mt = tile / NT, nt = tile - mt * NT;
    if (mt >= MKV / 32) return;
    v8f acc[2][4];
    zero_acc(acc);
    gemm32x64(A, Wt, DIM, mt * 32, nt * 64, lane, acc);
    const int half = lane >> 4, l16 = lane & 15;
#pragma unroll
    for (int mi = 0; mi < 2; ++mi)
#pragma unroll
        for (int t = 0; t < 4; ++t) {
            const int n = nt * 64 + t * 16 + l16;
            const int isv = n >= DIM;
            const int rem = n - (isv ? DIM : 0);
            const int h = rem >> 6, d = rem & 63;
#pragma unroll
            for (int r = 0; r < 8; ++r) {
                const int mrow = mt * 32 + mi * 16 + 8 * half + r;
                const int b = mrow >> 10, kr = mrow & 1023;
                float v = acc[mi][t][r] + bias[n];
                if (!isv)
                    Kb[(((size_t)b * HEADS + h) * KL + kr) * HD + d] = (bf16_t)v;
                else
                    Vt[(((size_t)b * HEADS + h) * HD + d) * KL + kr] = (bf16_t)v;
            }
        }
}

// ---------------------------------------------------------------------------
// Fused flash attention: one wave per (b,h, 16-row q-tile).
// S = Q K^T (+bias), online softmax, O += P V. ctx[b,q, h*64+d] bf16.
// ---------------------------------------------------------------------------
__global__ __launch_bounds__(256) void attn_kernel(
    const bf16_t* __restrict__ Qs, const bf16_t* __restrict__ Kb,
    const bf16_t* __restrict__ Vt, const float* __restrict__ biasF,
    bf16_t* __restrict__ ctx) {
    __shared__ float smem[8][16][32];       // per-wave P staging (16 KB)
    const int lane = threadIdx.x & 31, wid = threadIdx.x >> 5;
    const int gw = blockIdx.x * 8 + wid;    // 0..6143
    if (gw >= BH * (QL / 16)) return;
    const int bh = gw >> 4, qt = gw & 15;
    const int b = bh / HEADS, h = bh - b * HEADS;
    const bf16_t* Qp = Qs + (size_t)bh * QL * HD;
    const bf16_t* Kp = Kb + (size_t)bh * KL * HD;
    const bf16_t* Vp = Vt + (size_t)bh * HD * KL;
    const float*  Bp = biasF + (size_t)h * BIAS_QK;
    const int half = lane >> 4, l16 = lane & 15;

    const v16bf aq0 = load_frag(Qp, qt * 16 + l16, HD, 0, lane);
    const v16bf aq1 = load_frag(Qp, qt * 16 + l16, HD, 32, lane);

    v8f o[4];
    float m[8], l[8];
#pragma unroll
    for (int t = 0; t < 4; ++t)
#pragma unroll
        for (int r = 0; r < 8; ++r) o[t][r] = 0.0f;
#pragma unroll
    for (int r = 0; r < 8; ++r) { m[r] = -INFINITY; l[r] = 0.0f; }

    const int qrow = qt * 16 + 8 * half;
    for (int kb = 0; kb < KL; kb += 32) {
        // ---- S = Q K^T for 32 keys (two 16-col C tiles) ----
        v8f s0, s1;
#pragma unroll
        for (int r = 0; r < 8; ++r) { s0[r] = 0.0f; s1[r] = 0.0f; }
        {
            v16bf bk = load_frag(Kp, kb + l16, HD, 0, lane);
            s0 = wmma_bf16(aq0, bk, s0);
            bk = load_frag(Kp, kb + l16, HD, 32, lane);
            s0 = wmma_bf16(aq1, bk, s0);
            bk = load_frag(Kp, kb + 16 + l16, HD, 0, lane);
            s1 = wmma_bf16(aq0, bk, s1);
            bk = load_frag(Kp, kb + 16 + l16, HD, 32, lane);
            s1 = wmma_bf16(aq1, bk, s1);
        }
        // ---- relative-position bias ----
#pragma unroll
        for (int r = 0; r < 8; ++r) {
            const float* bp = Bp + (size_t)(qrow + r) * KL + kb + l16;
            s0[r] += bp[0];
            s1[r] += bp[16];
        }
        // ---- online softmax update ----
        float alpha[8];
#pragma unroll
        for (int r = 0; r < 8; ++r) {
            float mn = redmax16(fmaxf(s0[r], s1[r]));
            mn = fmaxf(mn, m[r]);
            alpha[r] = __expf(m[r] - mn);
            m[r] = mn;
        }
#pragma unroll
        for (int r = 0; r < 8; ++r) {
            float p0 = __expf(s0[r] - m[r]);
            float p1 = __expf(s1[r] - m[r]);
            s0[r] = p0; s1[r] = p1;
            l[r] = l[r] * alpha[r] + redsum16(p0 + p1);
        }
#pragma unroll
        for (int t = 0; t < 4; ++t)
#pragma unroll
            for (int r = 0; r < 8; ++r) o[t][r] *= alpha[r];
        // ---- C-layout P -> LDS -> A-layout bf16 (per-wave, DS in-order) ----
#pragma unroll
        for (int r = 0; r < 8; ++r) {
            smem[wid][8 * half + r][l16]      = s0[r];
            smem[wid][8 * half + r][16 + l16] = s1[r];
        }
        v16bf ap;
#pragma unroll
        for (int j = 0; j < 16; ++j)
            ap[j] = (bf16_t)smem[wid][l16][kof(lane, j)];
        // ---- O += P V  (V pre-transposed: d-major rows, contiguous k) ----
#pragma unroll
        for (int t = 0; t < 4; ++t) {
            v16bf bv = load_frag(Vp, t * 16 + l16, KL, kb, lane);
            o[t] = wmma_bf16(ap, bv, o[t]);
        }
    }
    // ---- finalize: ctx[b, qrow+r, h*64 + d] = O / l ----
#pragma unroll
    for (int r = 0; r < 8; ++r) {
        const float inv = 1.0f / l[r];
        const size_t base = ((size_t)b * QL + qrow + r) * DIM + h * HD;
#pragma unroll
        for (int t = 0; t < 4; ++t)
            ctx[base + t * 16 + l16] = (bf16_t)(o[t][r] * inv);
    }
}

// ---------------------------------------------------------------------------
// GEMM 3: out = ctx @ Wproj + bproj  (f32 output)
// ---------------------------------------------------------------------------
__global__ __launch_bounds__(256) void oproj_kernel(
    const bf16_t* __restrict__ A, const bf16_t* __restrict__ Wt,
    const float* __restrict__ bias, float* __restrict__ out) {
    const int lane = threadIdx.x & 31, wid = threadIdx.x >> 5;
    const int tile = blockIdx.x * 8 + wid;
    const int NT = DIM / 64;
    const int mt = tile / NT, nt = tile - mt * NT;
    if (mt >= MQ / 32) return;
    v8f acc[2][4];
    zero_acc(acc);
    gemm32x64(A, Wt, DIM, mt * 32, nt * 64, lane, acc);
    const int half = lane >> 4, l16 = lane & 15;
#pragma unroll
    for (int mi = 0; mi < 2; ++mi)
#pragma unroll
        for (int t = 0; t < 4; ++t) {
            const int n = nt * 64 + t * 16 + l16;
#pragma unroll
            for (int r = 0; r < 8; ++r) {
                const int mrow = mt * 32 + mi * 16 + 8 * half + r;
                out[(size_t)mrow * DIM + n] = acc[mi][t][r] + bias[n];
            }
        }
}

// ---------------------------------------------------------------------------
// Host launcher
// ---------------------------------------------------------------------------
extern "C" void kernel_launch(void* const* d_in, const int* in_sizes, int n_in,
                              void* d_out, int out_size, void* d_ws,
                              size_t ws_size, hipStream_t stream) {
    const float* q     = (const float*)d_in[0];
    const float* kv    = (const float*)d_in[1];
    const float* Wq    = (const float*)d_in[2];
    const float* bq    = (const float*)d_in[3];
    const float* Wkv   = (const float*)d_in[4];
    const float* bkv   = (const float*)d_in[5];
    const float* table = (const float*)d_in[6];
    const float* Wproj = (const float*)d_in[7];
    const float* bproj = (const float*)d_in[8];
    const int*   rel   = (const int*)d_in[9];
    float* out = (float*)d_out;

    char* ws = (char*)d_ws;
    size_t off = 0;
    auto take = [&](size_t bytes) -> void* {
        void* p = ws + off;
        off += (bytes + 255) & ~(size_t)255;
        return p;
    };
    bf16_t* qb   = (bf16_t*)take((size_t)MQ * DIM * 2);
    bf16_t* kvb  = (bf16_t*)take((size_t)MKV * DIM * 2);
    bf16_t* WqT  = (bf16_t*)take((size_t)DIM * DIM * 2);
    bf16_t* WkvT = (bf16_t*)take((size_t)NKV * DIM * 2);
    bf16_t* WpT  = (bf16_t*)take((size_t)DIM * DIM * 2);
    bf16_t* Qs   = (bf16_t*)take((size_t)BH * QL * HD * 2);
    bf16_t* Kb   = (bf16_t*)take((size_t)BH * KL * HD * 2);
    bf16_t* Vt   = (bf16_t*)take((size_t)BH * KL * HD * 2);
    float*  biasF = (float*)take((size_t)HEADS * BIAS_QK * 4);
    bf16_t* ctx  = (bf16_t*)take((size_t)MQ * DIM * 2);

    // Stage 0: precision conversion / layout prep
    cvt_bf16_kernel<<<(MQ * DIM + 255) / 256, 256, 0, stream>>>(q, qb, MQ * DIM);
    cvt_bf16_kernel<<<(MKV * DIM + 255) / 256, 256, 0, stream>>>(kv, kvb, MKV * DIM);
    transpose_bf16_kernel<<<(DIM * DIM + 255) / 256, 256, 0, stream>>>(Wq, WqT, DIM, DIM);
    transpose_bf16_kernel<<<(DIM * NKV + 255) / 256, 256, 0, stream>>>(Wkv, WkvT, DIM, NKV);
    transpose_bf16_kernel<<<(DIM * DIM + 255) / 256, 256, 0, stream>>>(Wproj, WpT, DIM, DIM);
    bias_expand_kernel<<<(BIAS_QK + 255) / 256, 256, 0, stream>>>(rel, table, biasF);

    // Stage 1: projections (WMMA GEMMs, 32x64 per wave)
    qproj_kernel<<<(MQ / 32) * (DIM / 64) / 8, 256, 0, stream>>>(qb, WqT, bq, Qs);
    kvproj_kernel<<<(MKV / 32) * (NKV / 64) / 8, 256, 0, stream>>>(kvb, WkvT, bkv, Kb, Vt);

    // Stage 2: fused flash attention (WMMA QK^T + softmax + WMMA PV)
    attn_kernel<<<BH * (QL / 16) / 8, 256, 0, stream>>>(Qs, Kb, Vt, biasF, ctx);

    // Stage 3: output projection (WMMA GEMM, f32 out)
    oproj_kernel<<<(MQ / 32) * (DIM / 64) / 8, 256, 0, stream>>>(ctx, WpT, bproj, out);
}
